// LSTMGenerator_Diffusion_84198538870824
// MI455X (gfx1250) — compile-verified, matrix operand order checked by reading, not run on hardware
//
#include <hip/hip_runtime.h>
#include <math.h>

// ---------------------------------------------------------------------------
// LSTM diffusion generator for MI455X (gfx1250, wave32, WMMA).
// L=2 => the diffusion loop body runs exactly once and no RNG is needed.
// All big GEMMs run as bf16 v_wmma_f32_16x16x32_bf16 with LDS double buffering.
// Round 2: per-mode templated epilogues + cheap transcendentals + relaxed
// VGPR budget to eliminate the accumulator scratch-spill seen in round 1.
// ---------------------------------------------------------------------------

typedef __attribute__((ext_vector_type(16))) __bf16 v16bf;
typedef __attribute__((ext_vector_type(8)))  __bf16 v8bf;
typedef __attribute__((ext_vector_type(8)))  float  v8f;
typedef __bf16 bf16_t;

#define NB      32768        // batch B
#define HID     1024
#define TSTEPS  128

// register-light tanh/sigmoid: one v_exp_f32 + one v_rcp_f32 each
__device__ __forceinline__ float fast_tanh(float x) {
    x = fminf(15.f, fmaxf(-15.f, x));
    const float e = __expf(2.f * x);
    return (e - 1.f) * __builtin_amdgcn_rcpf(e + 1.f);
}
__device__ __forceinline__ float fast_sig(float x) {
    x = fminf(30.f, fmaxf(-30.f, x));
    return __builtin_amdgcn_rcpf(1.f + __expf(-x));
}

// ---------------- GEMM: C(MxN) = A(MxK) @ Bw(NxK)^T, bf16 in, f32 acc -------
#define BM 128
#define BN 128
#define BK 32
#define LPITCH 40            // halves per LDS row (pad vs bank conflicts, 16B-aligned)

// epilogue modes
#define EP_LEAKY_RES   0     // leaky(v+bias)+resid -> bf16 outB0
#define EP_TANH_SPLIT  1     // tanh(v+bias) -> bf16: n<1024 -> outB0[m*1024+n], else outB1[m*2048+n]
#define EP_TANH_F32    2     // tanh(v+bias) -> f32 outF
#define EP_LSTM0       3     // v + zx[b,:]@wi0[n,:] + bi0[n]+bh0[n] -> f32 outF
#define EP_BIAS2       4     // v + bias[n] + bias2[n] -> f32 outF

template<int MODE>
__global__ __launch_bounds__(256, 2) void gemm_bf16_wmma(
    const bf16_t* __restrict__ A, const bf16_t* __restrict__ Bw,
    int M, int N, int K,
    const float* __restrict__ bias, const float* __restrict__ bias2,
    const float* __restrict__ leak, const bf16_t* __restrict__ resid,
    const float* __restrict__ wi0,  const float* __restrict__ zx,
    float* __restrict__ outF, bf16_t* __restrict__ outB0, bf16_t* __restrict__ outB1)
{
    __shared__ bf16_t lA[2][BM * LPITCH];
    __shared__ bf16_t lB[2][BN * LPITCH];

    const int tid   = threadIdx.x;
    const int m0    = blockIdx.y * BM;
    const int n0    = blockIdx.x * BN;
    const int wave  = tid >> 5;          // 8 waves
    const int lane  = tid & 31;
    const int waveM = wave & 3;          // 4 wave-rows  x 32 rows
    const int waveN = wave >> 2;         // 2 wave-cols  x 64 cols
    const int ln    = lane & 15;
    const int hi    = lane >> 4;         // half-wave select (K interleave per ISA A layout)

    // global->LDS: each of 256 threads moves 16 halves (32B) of A and of B per k-step
    const int grow  = tid >> 1;          // 0..127
    const int ghalf = tid & 1;           // 0..1 (16-half chunk)

    const v8f vzero = {0.f,0.f,0.f,0.f,0.f,0.f,0.f,0.f};
    v8f acc[2][4];
#pragma unroll
    for (int i = 0; i < 2; ++i)
#pragma unroll
        for (int j = 0; j < 4; ++j) acc[i][j] = vzero;

    // preload k-step 0 into stage 0
    {
        const uint4* pa = (const uint4*)(A  + (size_t)(m0 + grow) * K + ghalf * 16);
        const uint4* pb = (const uint4*)(Bw + (size_t)(n0 + grow) * K + ghalf * 16);
        uint4 ra0 = pa[0], ra1 = pa[1];
        uint4 rb0 = pb[0], rb1 = pb[1];
        uint4* da = (uint4*)&lA[0][grow * LPITCH + ghalf * 16];
        uint4* db = (uint4*)&lB[0][grow * LPITCH + ghalf * 16];
        da[0] = ra0; da[1] = ra1;
        db[0] = rb0; db[1] = rb1;
    }

    const int nk = K / BK;
    int cur = 0;
    for (int ks = 0; ks < nk; ++ks) {
        uint4 ra0, ra1, rb0, rb1;
        const bool more = (ks + 1) < nk;
        if (more) {
            const int kk = (ks + 1) * BK;
            const uint4* pa = (const uint4*)(A  + (size_t)(m0 + grow) * K + kk + ghalf * 16);
            const uint4* pb = (const uint4*)(Bw + (size_t)(n0 + grow) * K + kk + ghalf * 16);
            ra0 = pa[0]; ra1 = pa[1];
            rb0 = pb[0]; rb1 = pb[1];
        }
        __syncthreads();

        // fragment loads from LDS (16-bit A-layout: lane<16 holds K 0-7/16-23, lane>=16 K 8-15/24-31)
        v16bf afrag[2], bfrag[4];
#pragma unroll
        for (int mf = 0; mf < 2; ++mf) {
            const int row = waveM * 32 + mf * 16 + ln;
            const v8bf lo = *(const v8bf*)&lA[cur][row * LPITCH + hi * 8];
            const v8bf hv = *(const v8bf*)&lA[cur][row * LPITCH + 16 + hi * 8];
            v16bf f;
#pragma unroll
            for (int i = 0; i < 8; ++i) { f[i] = lo[i]; f[i + 8] = hv[i]; }
            afrag[mf] = f;
        }
#pragma unroll
        for (int nf = 0; nf < 4; ++nf) {
            const int row = waveN * 64 + nf * 16 + ln;
            const v8bf lo = *(const v8bf*)&lB[cur][row * LPITCH + hi * 8];
            const v8bf hv = *(const v8bf*)&lB[cur][row * LPITCH + 16 + hi * 8];
            v16bf f;
#pragma unroll
            for (int i = 0; i < 8; ++i) { f[i] = lo[i]; f[i + 8] = hv[i]; }
            bfrag[nf] = f;
        }
#pragma unroll
        for (int mf = 0; mf < 2; ++mf)
#pragma unroll
            for (int nf = 0; nf < 4; ++nf)
                acc[mf][nf] = __builtin_amdgcn_wmma_f32_16x16x32_bf16(
                    false, afrag[mf], false, bfrag[nf], (short)0, acc[mf][nf], false, false);

        __syncthreads();
        if (more) {
            const int nxt = cur ^ 1;
            uint4* da = (uint4*)&lA[nxt][grow * LPITCH + ghalf * 16];
            uint4* db = (uint4*)&lB[nxt][grow * LPITCH + ghalf * 16];
            da[0] = ra0; da[1] = ra1;
            db[0] = rb0; db[1] = rb1;
            cur = nxt;
        }
    }

    // epilogue: C/D layout: VGPR r -> lanes0-15: (M=r, N=lane); lanes16-31: (M=r+8, N=lane-16)
    float al = 0.f;
    if (MODE == EP_LEAKY_RES) al = leak[0];
#pragma unroll
    for (int mf = 0; mf < 2; ++mf)
        for (int nf = 0; nf < 4; ++nf)
#pragma unroll
            for (int r = 0; r < 8; ++r) {
                const int m = m0 + waveM * 32 + mf * 16 + r + hi * 8;
                const int n = n0 + waveN * 64 + nf * 16 + ln;
                const float v = acc[mf][nf][r];
                const size_t idx = (size_t)m * N + n;
                if (MODE == EP_LEAKY_RES) {
                    float t = v + bias[n];
                    t = t > 0.f ? t : al * t;
                    t += (float)resid[idx];
                    outB0[idx] = (bf16_t)t;
                } else if (MODE == EP_TANH_SPLIT) {
                    const float t = fast_tanh(v + bias[n]);
                    if (n < HID) outB0[(size_t)m * HID + n] = (bf16_t)t;
                    else         outB1[(size_t)m * (2 * HID) + n] = (bf16_t)t;
                } else if (MODE == EP_TANH_F32) {
                    outF[idx] = fast_tanh(v + bias[n]);
                } else if (MODE == EP_LSTM0) {
                    outF[idx] = v + zx[m * 2] * wi0[n * 2] + zx[m * 2 + 1] * wi0[n * 2 + 1]
                                  + bias[n] + bias2[n];
                } else { // EP_BIAS2
                    outF[idx] = v + bias[n] + bias2[n];
                }
            }
}

// ---------------- elementwise kernels --------------------------------------

// zero the (T,B,2,1) output; seed slice t=127 with z=ns[:,:,0]; also stash z into zx (B*2 f32)
__global__ void init_out_kernel(float* __restrict__ out, const float* __restrict__ ns,
                                float* __restrict__ zx, int total, int BL)
{
    const int idx = blockIdx.x * blockDim.x + threadIdx.x;
    if (idx >= total) return;
    const int t = idx / BL;
    const int rem = idx - t * BL;
    float v = 0.f;
    if (t == TSTEPS - 1) {
        const int b = rem >> 1, l = rem & 1;
        v = ns[b * 4 + l * 2];      // noise_start_seq_z[b, l, 0]
    }
    out[idx] = v;
    if (idx < BL) {
        const int b = idx >> 1, l = idx & 1;
        zx[idx] = ns[b * 4 + l * 2];
    }
}

__global__ void cvt_bf16(const float* __restrict__ src, bf16_t* __restrict__ dst, int n)
{
    const int i = blockIdx.x * blockDim.x + threadIdx.x;
    if (i < n) dst[i] = (bf16_t)src[i];
}

// wcat (4096 x 2048) = [wi1 | wh1] along K, bf16
__global__ void build_wcat(const float* __restrict__ wi1, const float* __restrict__ wh1,
                           bf16_t* __restrict__ wcat)
{
    const int idx = blockIdx.x * blockDim.x + threadIdx.x; // 4096*2048
    const int n = idx >> 11;
    const int k = idx & 2047;
    const float v = (k < HID) ? wi1[n * HID + k] : wh1[n * HID + (k - HID)];
    wcat[idx] = (bf16_t)v;
}

// y0[b,n] = leaky(0.3*ni[b,:] @ w0[n,:] + b0[n]) -> bf16   (K=2 input FC of resnet)
__global__ void fc_k2_leaky(const float* __restrict__ ni, const float* __restrict__ w0,
                            const float* __restrict__ b0, const float* __restrict__ a0,
                            bf16_t* __restrict__ y0)
{
    const int idx = blockIdx.x * blockDim.x + threadIdx.x;  // B*HID
    const int b = idx >> 10;
    const int n = idx & (HID - 1);
    const float x0 = 0.3f * ni[b * 2];
    const float x1 = 0.3f * ni[b * 2 + 1];
    float t = x0 * w0[n * 2] + x1 * w0[n * 2 + 1] + b0[n];
    t = t > 0.f ? t : a0[0] * t;
    y0[idx] = (bf16_t)t;
}

// LSTM gate math. layer 0: h_new -> bf16 into A1[:,0:1024]; layer 1: tanh(h_new) -> f32 th1
__global__ void lstm_pw(const float* __restrict__ g, const float* __restrict__ c0, int layer,
                        bf16_t* __restrict__ hOutB, float* __restrict__ hOutF)
{
    const int idx = blockIdx.x * blockDim.x + threadIdx.x;  // B*HID
    const int b = idx >> 10;
    const int n = idx & (HID - 1);
    const float* gr = g + (size_t)b * (4 * HID);
    const float i_ = gr[n];
    const float f_ = gr[HID + n];
    const float g_ = gr[2 * HID + n];
    const float o_ = gr[3 * HID + n];
    const float c  = c0[(size_t)b * (2 * HID) + layer * HID + n];
    const float cn = fast_sig(f_) * c + fast_sig(i_) * fast_tanh(g_);
    const float hn = fast_sig(o_) * fast_tanh(cn);
    if (layer == 0) hOutB[(size_t)b * (2 * HID) + n] = (bf16_t)hn;
    else            hOutF[idx] = fast_tanh(hn);
}

// decoded[b] = th1[b,:] @ dec_w ; out[126,b,l] = (z[b,l] - (1-a)/sqrt(1-ba)*decoded[b]) / sqrt(a)
__global__ void decode_update(const float* __restrict__ th1, const float* __restrict__ decw,
                              const float* __restrict__ alphas, const float* __restrict__ baralphas,
                              const float* __restrict__ zx, float* __restrict__ out)
{
    const int gwave = (blockIdx.x * blockDim.x + threadIdx.x) >> 5;  // one wave per row b
    const int lane  = threadIdx.x & 31;
    if (gwave >= NB) return;
    const float* row = th1 + (size_t)gwave * HID;
    float s = 0.f;
    for (int i = lane; i < HID; i += 32) s += row[i] * decw[i];
#pragma unroll
    for (int off = 16; off > 0; off >>= 1) s += __shfl_xor(s, off, 32);
    const float a = alphas[1], ba = baralphas[1];
    const float coef = (1.f - a) * rsqrtf(1.f - ba);
    const float inva = rsqrtf(a);
    if (lane < 2) {
        const int l = lane;
        out[(size_t)(TSTEPS - 2) * NB * 2 + gwave * 2 + l] = (zx[gwave * 2 + l] - coef * s) * inva;
    }
}

// ---------------- host launcher --------------------------------------------
// d_in index map (setup_inputs dict order; 'params' pytree-flattened, sorted keys):
//  0 noise_start_seq_z  1 noise_init  2 alphas  3 betas  4 baralphas
//  5 dec_w
//  6 init_c.a0  7 init_c.a1  8 init_c.b0  9 init_c.b1 10 init_c.b2
// 11 init_c.w0 12 init_c.w1 13 init_c.w2
// 14 init_h.a0 15 init_h.a1 16 init_h.b0 17 init_h.b1 18 init_h.b2
// 19 init_h.w0 20 init_h.w1 21 init_h.w2
// 22 lstm0.bh 23 lstm0.bi 24 lstm0.wh 25 lstm0.wi
// 26 lstm1.bh 27 lstm1.bi 28 lstm1.wh 29 lstm1.wi
// 30 batch_size 31 seq_len 32 dim_seq 33 num_diffusion_steps

static constexpr size_t OFF_W1H  = 0;                               // 1024x1024 bf16
static constexpr size_t OFF_W2H  = OFF_W1H  + 2097152;              // 2048x1024 bf16
static constexpr size_t OFF_W1C  = OFF_W2H  + 4194304;
static constexpr size_t OFF_W2C  = OFF_W1C  + 2097152;
static constexpr size_t OFF_WH0  = OFF_W2C  + 4194304;              // 4096x1024 bf16
static constexpr size_t OFF_WCAT = OFF_WH0  + 8388608;              // 4096x2048 bf16
static constexpr size_t OFF_ZX   = OFF_WCAT + 16777216;             // B*2 f32
static constexpr size_t OFF_Y0   = OFF_ZX   + (size_t)NB*2*4;       // B*1024 bf16
static constexpr size_t OFF_Y1   = OFF_Y0   + (size_t)NB*HID*2;
static constexpr size_t OFF_H0L0 = OFF_Y1   + (size_t)NB*HID*2;     // B*1024 bf16
static constexpr size_t OFF_A1   = OFF_H0L0 + (size_t)NB*HID*2;     // B*2048 bf16
static constexpr size_t OFF_C0   = OFF_A1   + (size_t)NB*2*HID*2;   // B*2048 f32
static constexpr size_t OFF_G    = OFF_C0   + (size_t)NB*2*HID*4;   // B*4096 f32
static constexpr size_t OFF_TH1  = OFF_G    + (size_t)NB*4*HID*4;   // B*1024 f32

extern "C" void kernel_launch(void* const* d_in, const int* in_sizes, int n_in,
                              void* d_out, int out_size, void* d_ws, size_t ws_size,
                              hipStream_t stream)
{
    const float* ns    = (const float*)d_in[0];
    const float* ni    = (const float*)d_in[1];
    const float* alph  = (const float*)d_in[2];
    const float* balph = (const float*)d_in[4];
    const float* decw  = (const float*)d_in[5];
    const float* c_a0  = (const float*)d_in[6];
    const float* c_a1  = (const float*)d_in[7];
    const float* c_b0  = (const float*)d_in[8];
    const float* c_b1  = (const float*)d_in[9];
    const float* c_b2  = (const float*)d_in[10];
    const float* c_w0  = (const float*)d_in[11];
    const float* c_w1  = (const float*)d_in[12];
    const float* c_w2  = (const float*)d_in[13];
    const float* h_a0  = (const float*)d_in[14];
    const float* h_a1  = (const float*)d_in[15];
    const float* h_b0  = (const float*)d_in[16];
    const float* h_b1  = (const float*)d_in[17];
    const float* h_b2  = (const float*)d_in[18];
    const float* h_w0  = (const float*)d_in[19];
    const float* h_w1  = (const float*)d_in[20];
    const float* h_w2  = (const float*)d_in[21];
    const float* l0_bh = (const float*)d_in[22];
    const float* l0_bi = (const float*)d_in[23];
    const float* l0_wh = (const float*)d_in[24];
    const float* l0_wi = (const float*)d_in[25];
    const float* l1_bh = (const float*)d_in[26];
    const float* l1_bi = (const float*)d_in[27];
    const float* l1_wh = (const float*)d_in[28];
    const float* l1_wi = (const float*)d_in[29];

    char* ws = (char*)d_ws;
    bf16_t* W1H  = (bf16_t*)(ws + OFF_W1H);
    bf16_t* W2H  = (bf16_t*)(ws + OFF_W2H);
    bf16_t* W1C  = (bf16_t*)(ws + OFF_W1C);
    bf16_t* W2C  = (bf16_t*)(ws + OFF_W2C);
    bf16_t* WH0  = (bf16_t*)(ws + OFF_WH0);
    bf16_t* WCAT = (bf16_t*)(ws + OFF_WCAT);
    float*  ZX   = (float*) (ws + OFF_ZX);
    bf16_t* Y0   = (bf16_t*)(ws + OFF_Y0);
    bf16_t* Y1   = (bf16_t*)(ws + OFF_Y1);
    bf16_t* H0L0 = (bf16_t*)(ws + OFF_H0L0);
    bf16_t* A1   = (bf16_t*)(ws + OFF_A1);
    float*  C0   = (float*) (ws + OFF_C0);
    float*  G    = (float*) (ws + OFF_G);
    float*  TH1  = (float*) (ws + OFF_TH1);
    float*  out  = (float*)d_out;

    const dim3 blk(256);
    const int M = NB;

    // 0) zero output, seed slice 127 with z, stash zx
    const int total = TSTEPS * NB * 2;
    init_out_kernel<<<total / 256, blk, 0, stream>>>(out, ns, ZX, total, NB * 2);

    // 1) weight conversion f32 -> bf16 (weights stay L2-resident afterwards)
    cvt_bf16<<<1048576 / 256, blk, 0, stream>>>(h_w1, W1H, 1048576);
    cvt_bf16<<<2097152 / 256, blk, 0, stream>>>(h_w2, W2H, 2097152);
    cvt_bf16<<<1048576 / 256, blk, 0, stream>>>(c_w1, W1C, 1048576);
    cvt_bf16<<<2097152 / 256, blk, 0, stream>>>(c_w2, W2C, 2097152);
    cvt_bf16<<<4194304 / 256, blk, 0, stream>>>(l0_wh, WH0, 4194304);
    build_wcat<<<8388608 / 256, blk, 0, stream>>>(l1_wi, l1_wh, WCAT);

    // 2) resnet init_h : y0 -> y1 -> h0 (split: layer0 -> H0L0, layer1 -> A1 upper)
    fc_k2_leaky<<<(M * HID) / 256, blk, 0, stream>>>(ni, h_w0, h_b0, h_a0, Y0);
    {
        dim3 g(HID / BN, M / BM);
        gemm_bf16_wmma<EP_LEAKY_RES><<<g, blk, 0, stream>>>(Y0, W1H, M, HID, HID,
            h_b1, nullptr, h_a1, Y0, nullptr, nullptr, nullptr, Y1, nullptr);
    }
    {
        dim3 g((2 * HID) / BN, M / BM);
        gemm_bf16_wmma<EP_TANH_SPLIT><<<g, blk, 0, stream>>>(Y1, W2H, M, 2 * HID, HID,
            h_b2, nullptr, nullptr, nullptr, nullptr, nullptr, nullptr, H0L0, A1);
    }

    // 3) resnet init_c : -> C0 (f32, B x 2048)
    fc_k2_leaky<<<(M * HID) / 256, blk, 0, stream>>>(ni, c_w0, c_b0, c_a0, Y0);
    {
        dim3 g(HID / BN, M / BM);
        gemm_bf16_wmma<EP_LEAKY_RES><<<g, blk, 0, stream>>>(Y0, W1C, M, HID, HID,
            c_b1, nullptr, c_a1, Y0, nullptr, nullptr, nullptr, Y1, nullptr);
    }
    {
        dim3 g((2 * HID) / BN, M / BM);
        gemm_bf16_wmma<EP_TANH_F32><<<g, blk, 0, stream>>>(Y1, W2C, M, 2 * HID, HID,
            c_b2, nullptr, nullptr, nullptr, nullptr, nullptr, C0, nullptr, nullptr);
    }

    // 4) LSTM layer 0: g = h0_l0 @ wh0^T + (x@wi0^T + bi0 + bh0)   [x = z, K=2 folded into epilogue]
    {
        dim3 g((4 * HID) / BN, M / BM);
        gemm_bf16_wmma<EP_LSTM0><<<g, blk, 0, stream>>>(H0L0, WH0, M, 4 * HID, HID,
            l0_bi, l0_bh, nullptr, nullptr, l0_wi, ZX, G, nullptr, nullptr);
    }
    lstm_pw<<<(M * HID) / 256, blk, 0, stream>>>(G, C0, 0, A1, nullptr);  // h_new0 -> A1[:,0:1024]

    // 5) LSTM layer 1: g = [h_new0 | h0_l1] @ [wi1|wh1]^T + bi1 + bh1
    {
        dim3 g((4 * HID) / BN, M / BM);
        gemm_bf16_wmma<EP_BIAS2><<<g, blk, 0, stream>>>(A1, WCAT, M, 4 * HID, 2 * HID,
            l1_bi, l1_bh, nullptr, nullptr, nullptr, nullptr, G, nullptr, nullptr);
    }
    lstm_pw<<<(M * HID) / 256, blk, 0, stream>>>(G, C0, 1, nullptr, TH1);  // tanh(h_new1)

    // 6) decoder dot + diffusion update into slice 126
    decode_update<<<(NB * 32) / 256, blk, 0, stream>>>(TH1, decw, alph, balph, ZX, out);

    (void)in_sizes; (void)n_in; (void)out_size; (void)ws_size;
}
